// AttentionWrapper_11020886081822
// MI455X (gfx1250) — matrix operand order repaired and checked by reading, not compiled, not run on hardware
//
#include <hip/hip_runtime.h>

// ---------------------------------------------------------------------------
// CDNA5 (gfx1250) attention block: QKV proj + RoPE + flash attention (with
// post-softmax suppression) + output proj.
//   - All matmuls via v_wmma_f32_16x16x32_bf16 (wave32)
//   - bf16 Q/K tile staging via global_load_async_to_lds_b128 (ASYNCcnt)
// ---------------------------------------------------------------------------

typedef __attribute__((ext_vector_type(16))) __bf16 v16bf;
typedef __attribute__((ext_vector_type(8)))  float  v8f;

typedef unsigned int   u32;
typedef unsigned short u16;
typedef unsigned long long u64;

#define B_       2
#define S_       2048
#define HID_     4096
#define NH_      32
#define NKV_     8
#define HD_      128
#define GROUPS_  (NH_ / NKV_)
#define SCALE_   0.08838834764831845f   // 1/sqrt(128)
#define SUPPRESS_ 0.1f

__device__ __forceinline__ u16 f2bf(float f) {
  u32 u = __builtin_bit_cast(u32, f);
  u32 r = u + 0x7FFFu + ((u >> 16) & 1u);   // round-to-nearest-even
  return (u16)(r >> 16);
}
__device__ __forceinline__ float bf2f(u16 h) {
  u32 u = ((u32)h) << 16;
  return __builtin_bit_cast(float, u);
}

union Frag {                 // 16 bf16 = 8 dwords, matches v16bf operand
  v16bf v;
  u32   u[8];
  u16   s[16];
};

// ISA 7.12.2 layouts (wave32):
// A 16x32 bf16: lane half 0 holds K 0..7 (VGPR0-3) & 16..23 (VGPR4-7),
//               lane half 1 holds K 8..15 & 24..31; row M = lane%16.
__device__ __forceinline__ int kloA(int j, int half) {
  return 2 * (j & 3) + ((j >> 2) << 4) + 8 * half;
}
// B 32x16 bf16: lane half 0 holds K 0..7 & 16..23, half 1 holds K 8..15 & 24..31; col N=lane%16.
__device__ __forceinline__ int kloB(int j, int half) {
  return 2 * (j & 3) + ((j >> 2) << 4) + 8 * half;
}

// Async raw copy global(bf16) -> LDS, 16B per lane, ASYNCcnt-tracked.
__device__ __forceinline__ void async_b128(u32 lds_off, const void* gptr) {
  u64 ga = (u64)gptr;
  asm volatile("global_load_async_to_lds_b128 %0, %1, off"
               :: "v"(lds_off), "v"(ga) : "memory");
}
__device__ __forceinline__ void wait_async0() {
  asm volatile("s_wait_asynccnt 0x0" ::: "memory");
}

// ---------------------------------------------------------------------------
// Generic bf16 WMMA GEMM: C[M,N] = A[M,K] * W[N,K]^T
//   AT = float (convert) or u16 (already bf16); OT = u16 (bf16) or float.
//   Block: 128 threads (4 waves), 64x128 output tile; wave tile 32x64
//   (2x4 sub-tiles -> 8 WMMA per 32-wide K step, A frags reused 4x).
// ---------------------------------------------------------------------------
template <typename AT, typename OT>
__global__ __launch_bounds__(128) void gemm_bf16_wmma(
    const AT* __restrict__ A, const float* __restrict__ W, OT* __restrict__ C,
    int M, int N, int K) {
  __shared__ u16 As[64][34];    // even stride -> aligned u32 pair reads
  __shared__ u16 Bs[128][34];

  const int tid  = threadIdx.x;
  const int lane = tid & 31;
  const int w    = tid >> 5;
  const int half = lane >> 4;
  const int l16  = lane & 15;
  const int wm   = w >> 1, wn = w & 1;
  const int m0   = blockIdx.y * 64;
  const int n0   = blockIdx.x * 128;

  v8f acc[2][4];
  for (int i = 0; i < 2; i++)
    for (int j = 0; j < 4; j++)
      for (int r = 0; r < 8; r++) acc[i][j][r] = 0.f;

  const int lrow = tid >> 3;        // 0..15
  const int lcol = (tid & 7) * 4;   // 0..28

  for (int k0 = 0; k0 < K; k0 += 32) {
    // Stage A tile (64x32) with f32->bf16 conversion when needed.
    for (int p = 0; p < 4; p++) {
      int r = lrow + p * 16;
      const AT* src = A + (size_t)(m0 + r) * K + k0 + lcol;
      u16 t0, t1, t2, t3;
      if constexpr (sizeof(AT) == 4) {
        float4 fv = *(const float4*)src;
        t0 = f2bf(fv.x); t1 = f2bf(fv.y); t2 = f2bf(fv.z); t3 = f2bf(fv.w);
      } else {
        ushort4 sv = *(const ushort4*)src;
        t0 = sv.x; t1 = sv.y; t2 = sv.z; t3 = sv.w;
      }
      As[r][lcol] = t0; As[r][lcol + 1] = t1;
      As[r][lcol + 2] = t2; As[r][lcol + 3] = t3;
    }
    // Stage W tile (128x32), always f32->bf16.
    for (int p = 0; p < 8; p++) {
      int r = lrow + p * 16;
      float4 wv = *(const float4*)(W + (size_t)(n0 + r) * K + k0 + lcol);
      Bs[r][lcol] = f2bf(wv.x); Bs[r][lcol + 1] = f2bf(wv.y);
      Bs[r][lcol + 2] = f2bf(wv.z); Bs[r][lcol + 3] = f2bf(wv.w);
    }
    __syncthreads();

    Frag a[2], bfrag[4];
    for (int i = 0; i < 2; i++) {
      int row = wm * 32 + i * 16 + l16;
      for (int j = 0; j < 8; j++)
        a[i].u[j] = *(const u32*)&As[row][kloA(j, half)];
    }
    for (int i = 0; i < 4; i++) {
      int row = wn * 64 + i * 16 + l16;
      for (int j = 0; j < 8; j++)
        bfrag[i].u[j] = *(const u32*)&Bs[row][kloB(j, half)];
    }
    for (int i = 0; i < 2; i++)
      for (int j = 0; j < 4; j++)
        acc[i][j] = __builtin_amdgcn_wmma_f32_16x16x32_bf16(
            false, a[i].v, false, bfrag[j].v, (short)0, acc[i][j], false, false);
    __syncthreads();
  }

  // C layout: lane l, VGPR r -> (M = r + 8*(l/16), N = l%16)
  for (int i = 0; i < 2; i++) {
    int row0 = m0 + wm * 32 + i * 16 + 8 * half;
    for (int j = 0; j < 4; j++) {
      int col = n0 + wn * 64 + j * 16 + l16;
      OT* dst = C + (size_t)row0 * N + col;
      for (int r = 0; r < 8; r++) {
        float v = acc[i][j][r];
        if constexpr (sizeof(OT) == 4) dst[(size_t)r * N] = v;
        else                           dst[(size_t)r * N] = f2bf(v);
      }
    }
  }
}

// ---------------------------------------------------------------------------
// RoPE in-place on bf16 Q and K workspaces. One thread per rotation pair.
// ---------------------------------------------------------------------------
__global__ void rope_kernel(u16* __restrict__ qw, u16* __restrict__ kw,
                            const int* __restrict__ pos) {
  const int NHEAD = NH_ + NKV_;
  int t = blockIdx.x * blockDim.x + threadIdx.x;
  if (t >= B_ * S_ * NHEAD * 64) return;
  int j    = t & 63;
  int head = (t >> 6) % NHEAD;
  int s    = (t / (64 * NHEAD)) % S_;
  int b    = t / (64 * NHEAD * S_);

  float p    = (float)pos[b * S_ + s];
  float freq = p * __powf(10000.f, -(float)j * (1.f / 64.f));
  float sn, cs;
  __sincosf(freq, &sn, &cs);

  u16* ptr;
  if (head < NH_)
    ptr = qw + (size_t)(b * S_ + s) * (NH_ * HD_) + head * HD_;
  else
    ptr = kw + (size_t)(b * S_ + s) * (NKV_ * HD_) + (head - NH_) * HD_;

  float x1 = bf2f(ptr[j]), x2 = bf2f(ptr[j + 64]);
  ptr[j]      = f2bf(x1 * cs - x2 * sn);
  ptr[j + 64] = f2bf(x2 * cs + x1 * sn);
}

// ---------------------------------------------------------------------------
// Flash attention with causal mask + post-softmax suppression.
// Block = 4 waves = 64 query rows for one (b, h); wave owns 16 rows.
// Q and K tiles staged via global_load_async_to_lds_b128 (raw bf16 copy,
// 272B row stride keeps every 16B lane chunk aligned). V is transposed on
// the way into LDS (async can't transpose), P round-trips through LDS to
// relayout C-fragment -> A-fragment. Numerator (PV) uses suppressed
// probabilities, denominator raw row sums == softmax-then-multiply.
// ---------------------------------------------------------------------------
#define QS_STRIDE 136   // shorts; 272B = multiple of 16B for async chunks
#define KS_STRIDE 136

__global__ __launch_bounds__(128) void flash_attn_kernel(
    const u16* __restrict__ Q, const u16* __restrict__ K,
    const u16* __restrict__ V, u16* __restrict__ O,
    const int* __restrict__ tgt_pos, const int* __restrict__ subj) {
  __shared__ u16 Qs[64][QS_STRIDE];  // 64 q rows x 128 dims (+pad)
  __shared__ u16 Ks[32][KS_STRIDE];  // 32 keys  x 128 dims (+pad)
  __shared__ u16 Vt[128][34];        // transposed: dim x key
  __shared__ u16 Ps[4][16][34];      // per-wave P relayout buffer

  const int blk = blockIdx.x;
  const int qt  = blk & 31;          // S/64 = 32 query tiles
  const int h   = (blk >> 5) & 31;
  const int b   = blk >> 10;
  const int kvh = h / GROUPS_;

  const int tid  = threadIdx.x;
  const int lane = tid & 31;
  const int w    = tid >> 5;
  const int half = lane >> 4;
  const int l16  = lane & 15;
  const int q0   = qt * 64;
  const int qmax = q0 + 63;

  const int tgt = tgt_pos[b];
  int subs[8];
  for (int i = 0; i < 8; i++) subs[i] = subj[b * 8 + i];

  // ---- Stage Q tile asynchronously: wave w copies its own 16 rows.
  {
    const u16* qg = Q + (size_t)(b * S_ + q0) * (NH_ * HD_) + h * HD_;
    for (int i = 0; i < 8; i++) {
      int row = w * 16 + i * 2 + half;          // 2 rows per instruction
      const u16* gp = qg + (size_t)row * (NH_ * HD_) + l16 * 8;
      async_b128((u32)(size_t)&Qs[row][l16 * 8], gp);
    }
    wait_async0();
  }
  __syncthreads();

  // Q fragments for this wave's 16 rows, 4 K-chunks of 32.
  Frag qf[4];
  {
    int row = w * 16 + l16;
    for (int kc = 0; kc < 4; kc++)
      for (int j = 0; j < 8; j++)
        qf[kc].u[j] = *(const u32*)&Qs[row][kc * 32 + kloA(j, half)];
  }

  v8f o[8];
  for (int dt = 0; dt < 8; dt++)
    for (int r = 0; r < 8; r++) o[dt][r] = 0.f;

  float m_run[8], l_run[8];
  for (int r = 0; r < 8; r++) { m_run[r] = -3.0e38f; l_run[r] = 0.f; }

  const int qbase = q0 + w * 16 + 8 * half;   // row for accumulator slot r

  for (int kt = 0; kt <= qmax; kt += 32) {
    __syncthreads();
    // ---- Stage K tile asynchronously (wave w: rows w*8 .. w*8+7).
    {
      const u16* kg = K + (size_t)(b * S_ + kt) * (NKV_ * HD_) + kvh * HD_;
      for (int i = 0; i < 4; i++) {
        int row = w * 8 + i * 2 + half;
        const u16* gp = kg + (size_t)row * (NKV_ * HD_) + l16 * 8;
        async_b128((u32)(size_t)&Ks[row][l16 * 8], gp);
      }
    }
    // ---- Stage V tile transposed (through VGPRs).
    {
      const u32* vs = (const u32*)(V + (size_t)(b * S_ + kt) * (NKV_ * HD_) + kvh * HD_);
      for (int i = 0; i < 16; i++) {
        int idx = i * 128 + tid;     // 0..2047 uints
        int r = idx >> 6, c2 = idx & 63;
        u32 vv = vs[(size_t)r * (NKV_ * HD_ / 2) + c2];
        Vt[c2 * 2][r]     = (u16)(vv & 0xFFFFu);
        Vt[c2 * 2 + 1][r] = (u16)(vv >> 16);
      }
    }
    wait_async0();
    __syncthreads();

    // Scores: two 16x16 tiles (32 keys), contraction over HD=128.
    v8f s[2];
    for (int t = 0; t < 2; t++) {
      v8f c;
      for (int r = 0; r < 8; r++) c[r] = 0.f;
      int krow = t * 16 + l16;
      for (int kc = 0; kc < 4; kc++) {
        Frag bf;
        for (int j = 0; j < 8; j++)
          bf.u[j] = *(const u32*)&Ks[krow][kc * 32 + kloB(j, half)];
        c = __builtin_amdgcn_wmma_f32_16x16x32_bf16(
            false, qf[kc].v, false, bf.v, (short)0, c, false, false);
      }
      s[t] = c;
    }

    // Scale + causal mask.
    for (int t = 0; t < 2; t++) {
      int key = kt + t * 16 + l16;
      for (int r = 0; r < 8; r++) {
        float v = s[t][r] * SCALE_;
        s[t][r] = (key <= qbase + r) ? v : -1.0e30f;
      }
    }

    // Streaming softmax; rows live in a 16-lane half -> 4 shfl_xor hops.
    float p0[8], p1[8];
    for (int r = 0; r < 8; r++) {
      float tm = fmaxf(s[0][r], s[1][r]);
      tm = fmaxf(tm, __shfl_xor(tm, 1, 32));
      tm = fmaxf(tm, __shfl_xor(tm, 2, 32));
      tm = fmaxf(tm, __shfl_xor(tm, 4, 32));
      tm = fmaxf(tm, __shfl_xor(tm, 8, 32));
      float mnew = fmaxf(m_run[r], tm);
      float fac  = __expf(m_run[r] - mnew);
      m_run[r] = mnew;
      p0[r] = __expf(s[0][r] - mnew);
      p1[r] = __expf(s[1][r] - mnew);
      float rs = p0[r] + p1[r];
      rs += __shfl_xor(rs, 1, 32);
      rs += __shfl_xor(rs, 2, 32);
      rs += __shfl_xor(rs, 4, 32);
      rs += __shfl_xor(rs, 8, 32);
      l_run[r] = l_run[r] * fac + rs;
      for (int dt = 0; dt < 8; dt++) o[dt][r] *= fac;
    }

    // Suppression (numerator only) + relayout P (C-frag -> A-frag via LDS).
    for (int t = 0; t < 2; t++) {
      int key = kt + t * 16 + l16;
      bool insub = false;
      for (int i = 0; i < 8; i++) insub = insub || (subs[i] == key);
      for (int r = 0; r < 8; r++) {
        float p = (t == 0) ? p0[r] : p1[r];
        if (insub && (qbase + r) == tgt) p *= SUPPRESS_;
        Ps[w][r + 8 * half][t * 16 + l16] = f2bf(p);
      }
    }

    // PV: A = P (16x32), B = V^T (32x16 per d-tile).
    Frag pf;
    for (int j = 0; j < 8; j++)
      pf.u[j] = *(const u32*)&Ps[w][l16][kloA(j, half)];
    for (int dt = 0; dt < 8; dt++) {
      Frag vf;
      int drow = dt * 16 + l16;
      for (int j = 0; j < 8; j++)
        vf.u[j] = *(const u32*)&Vt[drow][kloB(j, half)];
      o[dt] = __builtin_amdgcn_wmma_f32_16x16x32_bf16(
          false, pf.v, false, vf.v, (short)0, o[dt], false, false);
    }
  }

  // Epilogue: one reciprocal per row, then normalize and store bf16.
  for (int r = 0; r < 8; r++) {
    float inv = __builtin_amdgcn_rcpf(l_run[r]);
    int q = qbase + r;
    u16* dst = O + (size_t)(b * S_ + q) * (NH_ * HD_) + h * HD_ + l16;
    for (int dt = 0; dt < 8; dt++)
      dst[dt * 16] = f2bf(o[dt][r] * inv);
  }
}

// ---------------------------------------------------------------------------
// Launch: QKV GEMMs -> RoPE -> flash attention -> output GEMM.
// Workspace (bf16): Q 32MB | K 8MB | V 8MB | attn 32MB  (~84MB total).
// attention_mask input unused (causal mask computed analytically).
// ---------------------------------------------------------------------------
extern "C" void kernel_launch(void* const* d_in, const int* in_sizes, int n_in,
                              void* d_out, int out_size, void* d_ws,
                              size_t ws_size, hipStream_t stream) {
  const float* hidden = (const float*)d_in[0];
  const float* Wq = (const float*)d_in[1];
  const float* Wk = (const float*)d_in[2];
  const float* Wv = (const float*)d_in[3];
  const float* Wo = (const float*)d_in[4];
  const int*  pos  = (const int*)d_in[6];
  const int*  tgt  = (const int*)d_in[7];
  const int*  subj = (const int*)d_in[8];
  (void)in_sizes; (void)n_in; (void)out_size; (void)ws_size;

  char* ws = (char*)d_ws;
  const size_t qb = (size_t)(B_ * S_) * (NH_ * HD_) * 2;   // 32 MB
  const size_t kb = (size_t)(B_ * S_) * (NKV_ * HD_) * 2;  // 8 MB
  u16* q_ws = (u16*)ws;
  u16* k_ws = (u16*)(ws + qb);
  u16* v_ws = (u16*)(ws + qb + kb);
  u16* a_ws = (u16*)(ws + qb + 2 * kb);
  float* out = (float*)d_out;

  const int M = B_ * S_;  // 4096
  dim3 blk(128);

  gemm_bf16_wmma<float, u16><<<dim3((NH_ * HD_) / 128, M / 64), blk, 0, stream>>>(
      hidden, Wq, q_ws, M, NH_ * HD_, HID_);
  gemm_bf16_wmma<float, u16><<<dim3((NKV_ * HD_) / 128, M / 64), blk, 0, stream>>>(
      hidden, Wk, k_ws, M, NKV_ * HD_, HID_);
  gemm_bf16_wmma<float, u16><<<dim3((NKV_ * HD_) / 128, M / 64), blk, 0, stream>>>(
      hidden, Wv, v_ws, M, NKV_ * HD_, HID_);

  int rope_threads = B_ * S_ * (NH_ + NKV_) * 64;
  rope_kernel<<<(rope_threads + 255) / 256, 256, 0, stream>>>(q_ws, k_ws, pos);

  flash_attn_kernel<<<B_ * NH_ * (S_ / 64), 128, 0, stream>>>(
      q_ws, k_ws, v_ws, a_ws, tgt, subj);

  gemm_bf16_wmma<u16, float><<<dim3(HID_ / 128, M / 64), blk, 0, stream>>>(
      a_ws, Wo, out, M, HID_, NH_ * HD_);
}
